// SwitchMoE_15994458210491
// MI455X (gfx1250) — matrix operand order repaired and checked by reading, not compiled, not run on hardware
//
#include <hip/hip_runtime.h>
#include <hip/hip_bf16.h>
#include <math.h>

// Problem dims (reference: B=1, S=2048, D=768, H=1536, E=8, K=2)
#define S_TOK 2048
#define DDIM  768
#define HDIM  1536
#define NEXP  8
#define TOPK  2

typedef __attribute__((ext_vector_type(16))) __bf16 v16bf;
typedef __attribute__((ext_vector_type(2)))  __bf16 v2bf;
typedef __attribute__((ext_vector_type(8)))  float  v8f;

union AFrag {
  struct { uint4 lo; uint4 hi; } u;
  v16bf v;
};

__device__ __forceinline__ __bf16 f2bf(float f) { return (__bf16)f; }

// ---------------------------------------------------------------- init
__global__ void init_counts_kernel(int* counts) {
  if (threadIdx.x < NEXP) counts[threadIdx.x] = 0;
}

// ---------------------------------------------------------------- gating
__global__ __launch_bounds__(256) void gate_kernel(
    const float* __restrict__ x, const float* __restrict__ Wg,
    int* __restrict__ counts, int* __restrict__ idx_map,
    int* __restrict__ pos_map, float* __restrict__ probs) {
  int t = blockIdx.x * blockDim.x + threadIdx.x;
  if (t >= S_TOK) return;
  float s[NEXP];
#pragma unroll
  for (int e = 0; e < NEXP; ++e) s[e] = 0.0f;
  const float* xr = x + (size_t)t * DDIM;
  for (int d = 0; d < DDIM; d += 4) {
    float4 xv = *(const float4*)(xr + d);
    const float* w0 = Wg + (size_t)d * NEXP;
#pragma unroll
    for (int e = 0; e < NEXP; ++e) {
      s[e] += xv.x * w0[e] + xv.y * w0[NEXP + e] + xv.z * w0[2 * NEXP + e] +
              xv.w * w0[3 * NEXP + e];
    }
  }
  // top-2 with first-occurrence tie-break (matches jax.lax.top_k)
  int i0 = 0; float b0 = s[0];
#pragma unroll
  for (int e = 1; e < NEXP; ++e) if (s[e] > b0) { b0 = s[e]; i0 = e; }
  int i1 = -1; float b1v = -INFINITY;
#pragma unroll
  for (int e = 0; e < NEXP; ++e)
    if (e != i0 && s[e] > b1v) { b1v = s[e]; i1 = e; }
  float z  = expf(b1v - b0);
  float p0 = 1.0f / (1.0f + z);
  probs[2 * t]     = p0;
  probs[2 * t + 1] = 1.0f - p0;
  idx_map[2 * t]     = i0;
  idx_map[2 * t + 1] = i1;
  pos_map[2 * t]     = atomicAdd(&counts[i0], 1);
  pos_map[2 * t + 1] = atomicAdd(&counts[i1], 1);
}

// ---------------------------------------------------------------- scan
__global__ void scan_kernel(const int* __restrict__ counts, int* __restrict__ offsets) {
  if (threadIdx.x == 0 && blockIdx.x == 0) {
    int acc = 0;
    for (int e = 0; e < NEXP; ++e) { offsets[e] = acc; acc += counts[e]; }
  }
}

// ---------------------------------------------------------------- fill lists
__global__ __launch_bounds__(256) void fill_kernel(
    const int* __restrict__ idx_map, const int* __restrict__ pos_map,
    const int* __restrict__ offsets, int* __restrict__ tok_list,
    int* __restrict__ slot_map) {
  int i = blockIdx.x * blockDim.x + threadIdx.x;
  if (i >= S_TOK * TOPK) return;
  int e    = idx_map[i];
  int slot = offsets[e] + pos_map[i];
  tok_list[slot] = i >> 1;
  slot_map[i]    = slot;
}

// ---------------------------------------------------------------- x -> bf16
__global__ __launch_bounds__(256) void xconvert_kernel(
    const float* __restrict__ x, __bf16* __restrict__ xbf) {
  int i = (blockIdx.x * blockDim.x + threadIdx.x) * 8;
  if (i >= S_TOK * DDIM) return;
  float4 a = ((const float4*)(x + i))[0];
  float4 b = ((const float4*)(x + i))[1];
  __bf16 o[8] = {f2bf(a.x), f2bf(a.y), f2bf(a.z), f2bf(a.w),
                 f2bf(b.x), f2bf(b.y), f2bf(b.z), f2bf(b.w)};
  *(uint4*)(xbf + i) = *(const uint4*)o;
}

// ================================================================ GEMMs
// Block: 256 thr = 8 wave32. Tile M=64 x N=128, macro K-step 64 (2 x 32).
// Wave w: M-tile (w>>1), N-half (w&1): 8 v_wmma_f32_16x16x32_bf16 per K-step.
// A fragments: per-lane K-contiguous 16B halves -> direct global b128 loads.
// B fragments: fp32 weights converted->bf16, transposed into fragment layout
//              in LDS with packed 32-bit DS stores.

// h[slot, :] = gelu( gather(xbf)[slot, :] @ W1[e] + b1[e] )   (bf16 out)
__global__ __launch_bounds__(256) void moe_gemm1(
    const __bf16* __restrict__ xbf, const float* __restrict__ W1,
    const float* __restrict__ b1, const int* __restrict__ counts,
    const int* __restrict__ offsets, const int* __restrict__ tok_list,
    __bf16* __restrict__ hbuf) {
  const int e     = blockIdx.z;
  const int mbase = blockIdx.y * 64;
  const int cbase = blockIdx.x * 128;
  const int cnt   = counts[e];
  if (mbase >= cnt) return;  // block-uniform: EXEC all-1 for WMMA
  const int slotbase = offsets[e];

  const int tid  = threadIdx.x;
  const int lane = tid & 31;
  const int wv   = tid >> 5;
  const int wmt  = wv >> 1;
  const int wnh  = wv & 1;

  __shared__ __bf16 Bls[2][4096];  // 2 K-chunks x 8 N-tiles, fragment order
  __shared__ int    toks[64];

  if (tid < 64) {
    int m  = mbase + tid;
    int mc = (m < cnt) ? m : (cnt - 1);  // clamp: garbage rows masked on store
    toks[tid] = tok_list[slotbase + mc];
  }
  __syncthreads();

  // A: per-lane direct global fragment pointer (row fixed across K loop)
  const int tok = toks[wmt * 16 + (lane & 15)];
  const __bf16* aptr = xbf + (size_t)tok * DDIM + ((lane >> 4) << 3);

  // B staging plan: thread -> K-pair (2kc,2kc+1) x 8 columns
  const int kc    = tid >> 4;              // 0..15
  const int nsub  = tid & 15;              // n0 = nsub*8
  const int ntile = nsub >> 1;
  const int blane = (((kc >> 2) & 1) << 4) + ((nsub & 1) << 3);
  const int bi    = ((kc & 3) << 1) + ((kc >> 3) << 3);  // even fragment element index
  const float* __restrict__ Brow =
      W1 + (size_t)e * DDIM * HDIM + (size_t)(2 * kc) * HDIM + cbase + nsub * 8;

  v8f acc[4] = {};

  for (int k0 = 0; k0 < DDIM; k0 += 64) {
#pragma unroll
    for (int c = 0; c < 2; ++c) {  // stage B chunk c (K = k0+32c .. +31)
      const float* r0 = Brow + (size_t)(k0 + 32 * c) * HDIM;
      const float* r1 = r0 + HDIM;
      float4 p0 = ((const float4*)r0)[0], p1 = ((const float4*)r0)[1];
      float4 q0 = ((const float4*)r1)[0], q1 = ((const float4*)r1)[1];
      float lo[8] = {p0.x, p0.y, p0.z, p0.w, p1.x, p1.y, p1.z, p1.w};
      float hi[8] = {q0.x, q0.y, q0.z, q0.w, q1.x, q1.y, q1.z, q1.w};
      __bf16* dst = &Bls[c][ntile * 512 + blane * 16 + bi];
#pragma unroll
      for (int j8 = 0; j8 < 8; ++j8) {
        v2bf p; p.x = f2bf(lo[j8]); p.y = f2bf(hi[j8]);
        *(v2bf*)(dst + j8 * 16) = p;  // packed 32-bit DS store, 4B aligned
      }
    }
    if (k0 + 64 < DDIM) __builtin_prefetch(Brow + (size_t)(k0 + 64) * HDIM, 0, 1);
    __syncthreads();
#pragma unroll
    for (int c = 0; c < 2; ++c) {
      AFrag af;
      af.u.lo = *(const uint4*)(aptr + k0 + 32 * c);
      af.u.hi = *(const uint4*)(aptr + k0 + 32 * c + 16);
#pragma unroll
      for (int j = 0; j < 4; ++j) {
        v16bf b = *(const v16bf*)&Bls[c][((wnh * 4 + j) * 32 + lane) * 16];
        acc[j] = __builtin_amdgcn_wmma_f32_16x16x32_bf16(
            false, af.v, false, b, (short)0, acc[j], false, false);
      }
    }
    __syncthreads();
  }

  // epilogue: bias + exact GELU + bf16 store (masked rows)
#pragma unroll
  for (int j = 0; j < 4; ++j) {
    int col  = cbase + (wnh * 4 + j) * 16 + (lane & 15);
    float bb = b1[e * HDIM + col];
#pragma unroll
    for (int r = 0; r < 8; ++r) {
      int m = mbase + wmt * 16 + ((lane >> 4) << 3) + r;
      if (m < cnt) {
        float v = acc[j][r] + bb;
        v = 0.5f * v * (1.0f + erff(v * 0.70710678118654752f));
        hbuf[(size_t)(slotbase + m) * HDIM + col] = f2bf(v);
      }
    }
  }
}

// y[slot, :] = h[slot, :] @ W2[e] + b2[e]   (fp32 out, prob applied in combine)
__global__ __launch_bounds__(256) void moe_gemm2(
    const __bf16* __restrict__ hbuf, const float* __restrict__ W2,
    const float* __restrict__ b2, const int* __restrict__ counts,
    const int* __restrict__ offsets, float* __restrict__ ybuf) {
  const int e     = blockIdx.z;
  const int mbase = blockIdx.y * 64;
  const int cbase = blockIdx.x * 128;
  const int cnt   = counts[e];
  if (mbase >= cnt) return;
  const int slotbase = offsets[e];

  const int tid  = threadIdx.x;
  const int lane = tid & 31;
  const int wv   = tid >> 5;
  const int wmt  = wv >> 1;
  const int wnh  = wv & 1;

  __shared__ __bf16 Bls[2][4096];

  // A: clamped row, direct global fragment loads from hbuf
  int arow = mbase + wmt * 16 + (lane & 15);
  if (arow >= cnt) arow = cnt - 1;
  const __bf16* aptr =
      hbuf + (size_t)(slotbase + arow) * HDIM + ((lane >> 4) << 3);

  const int kc    = tid >> 4;
  const int nsub  = tid & 15;
  const int ntile = nsub >> 1;
  const int blane = (((kc >> 2) & 1) << 4) + ((nsub & 1) << 3);
  const int bi    = ((kc & 3) << 1) + ((kc >> 3) << 3);
  const float* __restrict__ Brow =
      W2 + (size_t)e * HDIM * DDIM + (size_t)(2 * kc) * DDIM + cbase + nsub * 8;

  v8f acc[4] = {};

  for (int k0 = 0; k0 < HDIM; k0 += 64) {
#pragma unroll
    for (int c = 0; c < 2; ++c) {
      const float* r0 = Brow + (size_t)(k0 + 32 * c) * DDIM;
      const float* r1 = r0 + DDIM;
      float4 p0 = ((const float4*)r0)[0], p1 = ((const float4*)r0)[1];
      float4 q0 = ((const float4*)r1)[0], q1 = ((const float4*)r1)[1];
      float lo[8] = {p0.x, p0.y, p0.z, p0.w, p1.x, p1.y, p1.z, p1.w};
      float hi[8] = {q0.x, q0.y, q0.z, q0.w, q1.x, q1.y, q1.z, q1.w};
      __bf16* dst = &Bls[c][ntile * 512 + blane * 16 + bi];
#pragma unroll
      for (int j8 = 0; j8 < 8; ++j8) {
        v2bf p; p.x = f2bf(lo[j8]); p.y = f2bf(hi[j8]);
        *(v2bf*)(dst + j8 * 16) = p;
      }
    }
    if (k0 + 64 < HDIM) __builtin_prefetch(Brow + (size_t)(k0 + 64) * DDIM, 0, 1);
    __syncthreads();
#pragma unroll
    for (int c = 0; c < 2; ++c) {
      AFrag af;
      af.u.lo = *(const uint4*)(aptr + k0 + 32 * c);
      af.u.hi = *(const uint4*)(aptr + k0 + 32 * c + 16);
#pragma unroll
      for (int j = 0; j < 4; ++j) {
        v16bf b = *(const v16bf*)&Bls[c][((wnh * 4 + j) * 32 + lane) * 16];
        acc[j] = __builtin_amdgcn_wmma_f32_16x16x32_bf16(
            false, af.v, false, b, (short)0, acc[j], false, false);
      }
    }
    __syncthreads();
  }

#pragma unroll
  for (int j = 0; j < 4; ++j) {
    int col  = cbase + (wnh * 4 + j) * 16 + (lane & 15);
    float bb = b2[e * DDIM + col];
#pragma unroll
    for (int r = 0; r < 8; ++r) {
      int m = mbase + wmt * 16 + ((lane >> 4) << 3) + r;
      if (m < cnt)
        ybuf[(size_t)(slotbase + m) * DDIM + col] = acc[j][r] + bb;
    }
  }
}

// ---------------------------------------------------------------- combine
// grid: (S_TOK, DDIM/256) -> no integer divide
__global__ __launch_bounds__(256) void combine_kernel(
    const float* __restrict__ ybuf, const float* __restrict__ probs,
    const int* __restrict__ slot_map, float* __restrict__ out) {
  int t = blockIdx.x;
  int d = blockIdx.y * 256 + threadIdx.x;
  float p0 = probs[2 * t], p1 = probs[2 * t + 1];
  int s0 = slot_map[2 * t], s1 = slot_map[2 * t + 1];
  out[(size_t)t * DDIM + d] =
      p0 * ybuf[(size_t)s0 * DDIM + d] + p1 * ybuf[(size_t)s1 * DDIM + d];
}

// ---------------------------------------------------------------- launch
extern "C" void kernel_launch(void* const* d_in, const int* in_sizes, int n_in,
                              void* d_out, int out_size, void* d_ws, size_t ws_size,
                              hipStream_t stream) {
  const float* x  = (const float*)d_in[0];
  const float* Wg = (const float*)d_in[1];
  const float* W1 = (const float*)d_in[2];
  const float* b1 = (const float*)d_in[3];
  const float* W2 = (const float*)d_in[4];
  const float* b2 = (const float*)d_in[5];
  float* out = (float*)d_out;

  char* w = (char*)d_ws;
  int*    counts   = (int*)(w);
  int*    offsets  = (int*)(w + 64);
  int*    idx_map  = (int*)(w + 128);
  int*    pos_map  = (int*)(w + 128 + 1 * 16384);
  int*    slot_map = (int*)(w + 128 + 2 * 16384);
  int*    tok_list = (int*)(w + 128 + 3 * 16384);
  float*  probs    = (float*)(w + 128 + 4 * 16384);
  __bf16* xbf      = (__bf16*)(w + 128 + 5 * 16384);                     // 3.0 MB
  __bf16* hbuf     = (__bf16*)(w + 128 + 5 * 16384 + 3145728);           // 12.6 MB
  float*  ybuf     = (float*)(w + 128 + 5 * 16384 + 3145728 + 12582912); // 12.6 MB

  init_counts_kernel<<<1, 32, 0, stream>>>(counts);
  gate_kernel<<<S_TOK / 256, 256, 0, stream>>>(x, Wg, counts, idx_map, pos_map, probs);
  scan_kernel<<<1, 1, 0, stream>>>(counts, offsets);
  fill_kernel<<<(S_TOK * TOPK) / 256, 256, 0, stream>>>(idx_map, pos_map, offsets,
                                                        tok_list, slot_map);
  xconvert_kernel<<<(S_TOK * DDIM / 8) / 256, 256, 0, stream>>>(x, xbf);

  dim3 g1(HDIM / 128, S_TOK / 64, NEXP);  // (12, 32, 8)
  moe_gemm1<<<g1, 256, 0, stream>>>(xbf, W1, b1, counts, offsets, tok_list, hbuf);

  dim3 g2(DDIM / 128, S_TOK / 64, NEXP);  // (6, 32, 8)
  moe_gemm2<<<g2, 256, 0, stream>>>(hbuf, W2, b2, counts, offsets, ybuf);

  dim3 gc(S_TOK, DDIM / 256);             // (2048, 3)
  combine_kernel<<<gc, 256, 0, stream>>>(ybuf, probs, slot_map, out);
}
